// SPVCNN_65893388255510
// MI455X (gfx1250) — matrix operand order repaired and checked by reading, not compiled, not run on hardware
//
#include <hip/hip_runtime.h>

// ---------------------------------------------------------------------------
// Types for CDNA5 WMMA / TDM (gfx1250, wave32)
// ---------------------------------------------------------------------------
typedef _Float16 h16;
typedef __attribute__((ext_vector_type(16))) _Float16 v16h;
typedef __attribute__((ext_vector_type(8)))  float    v8f;
typedef __attribute__((ext_vector_type(4)))  unsigned v4u;
typedef __attribute__((ext_vector_type(4)))  int      v4i;
typedef __attribute__((ext_vector_type(8)))  int      v8i;

union Frag16 { v16h v; unsigned u[8]; };

static inline unsigned blks(long n) { return (unsigned)((n + 255) / 256); }

// ---------------------------------------------------------------------------
// Tensor Data Mover: async 2D tile load (global -> LDS), D# per ISA 08 §8.
//   - group0: count=1, lds_addr, 57-bit global addr, type=2 ("image")
//   - group1: data_size=2B, tensor_dim0/1 = remaining extent (HW zero-fills
//             OOB), tile 64x64 elements, tensor_dim0_stride = row stride
// Issued by one wave per block; completion via s_wait_tensorcnt 0.
// ---------------------------------------------------------------------------
#if defined(__HIP_DEVICE_COMPILE__) && __has_builtin(__builtin_amdgcn_tensor_load_to_lds)
#define SPV_TDM 1
__device__ __forceinline__
void tdm_load_2d(void* ldsDst, const void* gsrc, int kElems, int nRows, int strideElems) {
  if (kElems < 0) kElems = 0;
  if (nRows  < 0) nRows  = 0;
  unsigned lds = (unsigned)(size_t)ldsDst;                  // LDS aperture: low 32 bits
  unsigned long long ga = (unsigned long long)(size_t)gsrc;
  v4u g0 = { 1u,                                            // count=1, user descriptor
             lds,
             (unsigned)ga,
             (unsigned)(((ga >> 32) & 0x01ffffffu) | (2u << 30)) };  // addr[56:32] | type=2
  unsigned uk = (unsigned)kElems, un = (unsigned)nRows, us = (unsigned)strideElems;
  v8i g1 = { (int)(1u << 16),                               // data_size = 1 -> 2 bytes
             (int)((uk & 0xffffu) << 16),                   // tensor_dim0[15:0]
             (int)(((uk >> 16) & 0xffffu) | ((un & 0xffffu) << 16)),   // dim0 hi | dim1 lo
             (int)(((un >> 16) & 0xffffu) | (64u << 16)),   // dim1 hi | tile_dim0 = 64
             (int)64,                                       // tile_dim1 = 64, tile_dim2 = 0
             (int)us,                                       // tensor_dim0_stride[31:0]
             0, 0 };
  v4i gz = { 0, 0, 0, 0 };
#if __clang_major__ >= 23
  v8i gz8 = { 0, 0, 0, 0, 0, 0, 0, 0 };
  __builtin_amdgcn_tensor_load_to_lds(g0, g1, gz, gz, gz8, 0);
#else
  __builtin_amdgcn_tensor_load_to_lds(g0, g1, gz, gz, 0);
#endif
}
#endif

// ---------------------------------------------------------------------------
// Generic implicit-GEMM convolution, WMMA f32 <- f16 x f16.
// Block: 256 threads = 8 waves as 4(M) x 2(N); block tile 64(M) x 64(N).
// Each wave owns a 16(M) x 32(N) register tile (2 f32 accumulators) and per
// K-chunk (64) issues 4 back-to-back v_wmma_f32_16x16x32_f16 for ILP.
//   MODE 0: 3x3x3 stride-1 SAME conv          (27 taps, grid g)
//   MODE 1: 1x1 conv / point linear            (flat rows; TDM-staged tiles)
//   MODE 2: 2x2x2 stride-2 VALID conv          (8 taps, in-grid 2g, out-grid g)
//   MODE 3: 2x2x2 stride-2 transposed conv     (parity tap = blockIdx.z,
//                                               rows = coarse grid g, out = 2g)
// Epilogue: y = (acc*scale + bias) * mask + residual, optional ReLU, f16 out.
// ---------------------------------------------------------------------------
template <int MODE>
__global__ __launch_bounds__(256)
void conv_wmma_t(const h16* __restrict__ in, const h16* __restrict__ wts,
                 const float* __restrict__ scale, const float* __restrict__ bias,
                 const h16* __restrict__ mask, const h16* __restrict__ resid,
                 h16* __restrict__ out,
                 int g, int M, int Cin, int Cout, int relu,
                 int istride, int ostride, int ooff)
{
  __shared__ h16 lA[64 * 64];   // 64 rows  x 64 k   (im2col tile)
  __shared__ h16 lB[64 * 64];   // 64 cout  x 64 k   (weight tile, k contiguous)

  const int tid  = threadIdx.x;
  const int lane = tid & 31;
  const int wave = tid >> 5;
  const int wm   = wave >> 1;          // 0..3 : 16-row M sub-tile
  const int wn   = wave & 1;           // 0..1 : 32-col N sub-tile
  const int m0   = blockIdx.x * 64;
  const int n0   = blockIdx.y * 64;
  const int tapP = blockIdx.z;         // deconv parity (MODE 3)
  const int Kt   = (MODE == 0) ? 27 : (MODE == 2 ? 8 : 1);
  const int Ktot = Kt * Cin;
  const int hid  = lane >> 4;
  const int l15  = lane & 15;

  v8f accv[2];
  accv[0] = (v8f){0.f, 0.f, 0.f, 0.f, 0.f, 0.f, 0.f, 0.f};
  accv[1] = (v8f){0.f, 0.f, 0.f, 0.f, 0.f, 0.f, 0.f, 0.f};

  for (int kc = 0; kc < Ktot; kc += 64) {
    bool manual = true;
#ifdef SPV_TDM
    if (MODE == 1) {
      // Both tiles are plain 2D sub-tiles of row-major matrices: use the TDM.
      if (wave == 0) {
        tdm_load_2d(lA, in  + (long)m0 * istride + kc, Ktot - kc, M    - m0, istride);
        tdm_load_2d(lB, wts + (long)n0 * Cin     + kc, Ktot - kc, Cout - n0, Cin);
        __builtin_amdgcn_s_wait_tensorcnt((short)0);
      }
      manual = false;
    }
#endif
    if (manual) {
      // ---- stage A tile: 64 rows x 64 k (halo gather / im2col) ----
      for (int i = tid; i < 64 * 64; i += 256) {
        int mr  = i >> 6;
        int kk  = (i & 63) + kc;
        int row = m0 + mr;
        h16 val = (h16)0.f;
        if (row < M && kk < Ktot) {
          if (MODE == 1 || MODE == 3) {
            val = in[(long)row * istride + kk];
          } else {
            int t = kk / Cin, c = kk - t * Cin;
            int z = row % g, y = (row / g) % g, x = row / (g * g);
            int xi, yi, zi, gi;
            if (MODE == 0) {
              xi = x + t / 9 - 1; yi = y + (t / 3) % 3 - 1; zi = z + t % 3 - 1; gi = g;
            } else { // MODE 2
              xi = 2 * x + ((t >> 2) & 1); yi = 2 * y + ((t >> 1) & 1); zi = 2 * z + (t & 1);
              gi = 2 * g;
            }
            if (xi >= 0 && xi < gi && yi >= 0 && yi < gi && zi >= 0 && zi < gi)
              val = in[((long)(xi * gi + yi) * gi + zi) * istride + c];
          }
        }
        lA[i] = val;
      }
      // ---- stage B tile: weights pre-packed as [tap][cout][cin] f16 ----
      for (int i = tid; i < 64 * 32; i += 256) {   // 2048 k-pairs
        int nn   = i >> 5;                         // 0..63
        int kp   = (i & 31) * 2;                   // 0..62
        int kk   = kc + kp;
        int cout = n0 + nn;
        unsigned pv = 0u;
        if (cout < Cout && kk < Ktot) {
          int t, c;
          if (MODE == 1)      { t = 0;    c = kk; }
          else if (MODE == 3) { t = tapP; c = kk; }
          else                { t = kk / Cin; c = kk - t * Cin; }
          pv = *(const unsigned*)(wts + ((long)t * Cout + cout) * Cin + c);
          if (kk + 1 >= Ktot) pv &= 0xffffu;
        }
        ((unsigned*)lB)[i] = pv;
      }
    }
    __syncthreads();

    // prefetch next weight chunk (lowered to global_prefetch_b8 on gfx1250)
    if (kc + 64 < Ktot)
      __builtin_prefetch(wts + (long)(kc + 64) * Cout, 0, 1);

    // ---- build fragments per documented 16x16x32 f16 layouts ----
    Frag16 a0, a1, b00, b01, b10, b11;
#pragma unroll
    for (int v = 0; v < 8; ++v) {
      int ka = (v < 4 ? 2 * v : 16 + 2 * (v - 4)) + 8 * hid;   // A: 16(M)x32(K)
      const h16* arow = &lA[(wm * 16 + l15) * 64];
      a0.u[v] = *(const unsigned*)&arow[ka];
      a1.u[v] = *(const unsigned*)&arow[32 + ka];
      int kb = (v < 4 ? 2 * v : 16 + 2 * (v - 4)) + 8 * hid;   // B: 32(K)x16(N)
      const h16* brow0 = &lB[(wn * 32 + l15) * 64];
      const h16* brow1 = &lB[(wn * 32 + 16 + l15) * 64];
      b00.u[v] = *(const unsigned*)&brow0[kb];
      b10.u[v] = *(const unsigned*)&brow0[32 + kb];
      b01.u[v] = *(const unsigned*)&brow1[kb];
      b11.u[v] = *(const unsigned*)&brow1[32 + kb];
    }
    accv[0] = __builtin_amdgcn_wmma_f32_16x16x32_f16(false, a0.v, false, b00.v,
                                                     (short)0, accv[0], false, false);
    accv[1] = __builtin_amdgcn_wmma_f32_16x16x32_f16(false, a0.v, false, b01.v,
                                                     (short)0, accv[1], false, false);
    accv[0] = __builtin_amdgcn_wmma_f32_16x16x32_f16(false, a1.v, false, b10.v,
                                                     (short)0, accv[0], false, false);
    accv[1] = __builtin_amdgcn_wmma_f32_16x16x32_f16(false, a1.v, false, b11.v,
                                                     (short)0, accv[1], false, false);
    __syncthreads();
  }

  // ---- fused epilogue: BN * mask (+ residual) (+ ReLU) -> f16 ----
#pragma unroll
  for (int ns = 0; ns < 2; ++ns) {
    int cout = n0 + wn * 32 + ns * 16 + l15;
    if (cout >= Cout) continue;
    float s  = scale ? scale[cout] : 1.f;
    float bb = bias  ? bias[cout]  : 0.f;
#pragma unroll
    for (int r = 0; r < 8; ++r) {
      int mloc = (lane < 16) ? r : (r + 8);
      int row  = m0 + wm * 16 + mloc;
      if (row >= M) continue;
      float v = accv[ns][r] * s + bb;
      long orow;
      if (MODE == 3) {
        int z = row % g, y = (row / g) % g, x = row / (g * g);
        int gf = 2 * g;
        orow = ((long)((2 * x + ((tapP >> 2) & 1)) * gf + (2 * y + ((tapP >> 1) & 1))) * gf
                + (2 * z + (tapP & 1)));
      } else {
        orow = row;
      }
      if (mask)  v *= (float)mask[orow];
      if (resid) v += (float)resid[(long)row * Cout + cout];
      if (relu && v < 0.f) v = 0.f;
      out[orow * (long)ostride + ooff + cout] = (h16)v;
    }
  }
}

// ---------------------------------------------------------------------------
// Support kernels
// ---------------------------------------------------------------------------
__global__ void k_fill_f32(float* p, long n) {
  long i = (long)blockIdx.x * blockDim.x + threadIdx.x;
  if (i < n) p[i] = 0.f;
}

// repack conv weights [k^3][cin][cout] f32 -> [tap][cout][cin] f16
__global__ void k_prep_w(const float* __restrict__ w, h16* __restrict__ o,
                         int Kt, int Cin, int Cout) {
  long i = (long)blockIdx.x * blockDim.x + threadIdx.x;
  long total = (long)Kt * Cin * Cout;
  if (i >= total) return;
  int cout = (int)(i % Cout);
  long r   = i / Cout;
  int cin  = (int)(r % Cin);
  int t    = (int)(r / Cin);
  o[((long)t * Cout + cout) * Cin + cin] = (h16)w[i];
}

// fold BatchNorm (+ optional linear bias) into per-channel scale/bias
__global__ void k_prep_bn(const float* g_, const float* b_, const float* m_,
                          const float* v_, const float* lb, float* sc, float* bi, int C) {
  int c = blockIdx.x * blockDim.x + threadIdx.x;
  if (c >= C) return;
  float s = g_[c] * rsqrtf(v_[c] + 1e-5f);
  float l = lb ? lb[c] : 0.f;
  sc[c] = s;
  bi[c] = b_[c] - m_[c] * s + l * s;
}

__global__ void k_vox_init_accum(const float* __restrict__ feats, const float* __restrict__ coords,
                                 float* acc, float* cnt, int Np, int g) {
  int p = blockIdx.x * blockDim.x + threadIdx.x;
  if (p >= Np) return;
  int ix = min(max((int)floorf(coords[p * 3 + 0]), 0), g - 1);
  int iy = min(max((int)floorf(coords[p * 3 + 1]), 0), g - 1);
  int iz = min(max((int)floorf(coords[p * 3 + 2]), 0), g - 1);
  long v = (long)(ix * g + iy) * g + iz;
  for (int c = 0; c < 4; ++c) atomicAdd(&acc[v * 4 + c], feats[p * 4 + c]);
  atomicAdd(&cnt[v], 1.f);
}

__global__ void k_vox_init_final(const float* acc, const float* cnt,
                                 h16* grid, h16* m0, long nvox) {
  long v = (long)blockIdx.x * blockDim.x + threadIdx.x;
  if (v >= nvox) return;
  float c = cnt[v];
  float d = fmaxf(c, 1.f);
  for (int ch = 0; ch < 4; ++ch) grid[v * 4 + ch] = (h16)(acc[v * 4 + ch] / d);
  m0[v] = (h16)(c > 0.f ? 1.f : 0.f);
}

__global__ void k_downmask(const h16* mi, h16* mo, int go) {
  long n = (long)go * go * go;
  long v = (long)blockIdx.x * blockDim.x + threadIdx.x;
  if (v >= n) return;
  int z = (int)(v % go), y = (int)((v / go) % go), x = (int)(v / ((long)go * go));
  int gi = 2 * go;
  float m = 0.f;
  for (int dx = 0; dx < 2; ++dx)
    for (int dy = 0; dy < 2; ++dy)
      for (int dz = 0; dz < 2; ++dz) {
        long u = (long)((2 * x + dx) * gi + (2 * y + dy)) * gi + (2 * z + dz);
        float t = (float)mi[u];
        if (t > m) m = t;
      }
  mo[v] = (h16)m;
}

__global__ void k_p2v_accum(const h16* __restrict__ pf, const float* __restrict__ coords,
                            float* acc, float* cnt, int Np, int C, int g, float invs) {
  long i = (long)blockIdx.x * blockDim.x + threadIdx.x;
  if (i >= (long)Np * C) return;
  int p = (int)(i / C), c = (int)(i % C);
  int ix = min(max((int)floorf(coords[p * 3 + 0] * invs), 0), g - 1);
  int iy = min(max((int)floorf(coords[p * 3 + 1] * invs), 0), g - 1);
  int iz = min(max((int)floorf(coords[p * 3 + 2] * invs), 0), g - 1);
  long v = (long)(ix * g + iy) * g + iz;
  atomicAdd(&acc[v * C + c], (float)pf[(long)p * C + c]);
  if (c == 0) atomicAdd(&cnt[v], 1.f);
}

__global__ void k_p2v_final(const float* acc, const float* cnt, const h16* mask,
                            h16* grid, long nvox, int C) {
  long i = (long)blockIdx.x * blockDim.x + threadIdx.x;
  if (i >= nvox * C) return;
  long v = i / C;
  float d = fmaxf(cnt[v], 1.f);
  grid[i] = (h16)(acc[i] / d * (float)mask[v]);
}

// trilinear devoxelize; optionally accumulates onto existing point features
__global__ void k_v2p(const h16* __restrict__ grid, const h16* __restrict__ mask,
                      const float* __restrict__ coords, h16* outp,
                      int Np, int C, int g, float invs, int accumulate) {
  long i = (long)blockIdx.x * blockDim.x + threadIdx.x;
  if (i >= (long)Np * C) return;
  int p = (int)(i / C), c = (int)(i % C);
  float px = coords[p * 3 + 0] * invs;
  float py = coords[p * 3 + 1] * invs;
  float pz = coords[p * 3 + 2] * invs;
  float bx = floorf(px), by = floorf(py), bz = floorf(pz);
  float fx = px - bx, fy = py - by, fz = pz - bz;
  int ix = (int)bx, iy = (int)by, iz = (int)bz;
  float r = accumulate ? (float)outp[i] : 0.f;
  for (int dx = 0; dx < 2; ++dx) {
    int xi = ix + dx; if (xi < 0 || xi >= g) continue;
    float wx = dx ? fx : 1.f - fx;
    for (int dy = 0; dy < 2; ++dy) {
      int yi = iy + dy; if (yi < 0 || yi >= g) continue;
      float wy = dy ? fy : 1.f - fy;
      for (int dz = 0; dz < 2; ++dz) {
        int zi = iz + dz; if (zi < 0 || zi >= g) continue;
        float wz = dz ? fz : 1.f - fz;
        long v = (long)(xi * g + yi) * g + zi;
        float w = wx * wy * wz * (float)mask[v];
        r += w * (float)grid[v * (long)C + c];
      }
    }
  }
  outp[i] = (h16)r;
}

__global__ void k_copy_ch(const h16* src, h16* dst, long nvox, int C,
                          int sstride, int soff, int dstride, int doff) {
  long i = (long)blockIdx.x * blockDim.x + threadIdx.x;
  if (i >= nvox * C) return;
  long v = i / C;
  int c  = (int)(i % C);
  dst[v * dstride + doff + c] = src[v * sstride + soff + c];
}

__global__ void k_h2f(const h16* src, float* dst, long n) {
  long i = (long)blockIdx.x * blockDim.x + threadIdx.x;
  if (i < n) dst[i] = (float)src[i];
}

// ---------------------------------------------------------------------------
// Host orchestration
// ---------------------------------------------------------------------------
extern "C" void kernel_launch(void* const* d_in, const int* in_sizes, int n_in,
                              void* d_out, int out_size, void* d_ws, size_t ws_size,
                              hipStream_t stream) {
  (void)in_sizes; (void)n_in; (void)out_size; (void)ws_size;
  const int  Np = 120000;
  const int  G  = 64;
  const long nv0 = (long)G * G * G;
  const long nv1 = nv0 / 8, nv2 = nv0 / 64, nv3 = nv0 / 512, nv4 = nv0 / 4096;

  const float* feats  = (const float*)d_in[0];
  const float* coords = (const float*)d_in[1];
  auto P = [&](int i) { return (const float*)d_in[i]; };

  // ---- parameter index walk, mirroring _make_params() insertion order ----
  int pi = 2;
  struct Layer { int w = -1, bn = -1; };
  auto takeConv = [&]() { Layer L; L.w = pi++; return L; };
  auto takeBN   = [&](Layer& L) { L.bn = pi; pi += 4; };
  struct Stage { Layer down, r1c1, r1c2, r1sc, r2c1, r2c2; bool hasSC; };
  auto takeStage = [&](bool sc) {
    Stage S; S.hasSC = sc;
    S.down = takeConv(); takeBN(S.down);
    S.r1c1 = takeConv(); takeBN(S.r1c1);
    S.r1c2 = takeConv(); takeBN(S.r1c2);
    if (sc) { S.r1sc = takeConv(); takeBN(S.r1sc); }
    S.r2c1 = takeConv(); takeBN(S.r2c1);
    S.r2c2 = takeConv(); takeBN(S.r2c2);
    return S;
  };
  Layer stem0 = takeConv(); takeBN(stem0);
  Layer stem1 = takeConv(); takeBN(stem1);
  Stage s1 = takeStage(false);   // 32->32
  Stage s2 = takeStage(true);    // 32->64
  Stage s3 = takeStage(true);    // 64->128
  Stage s4 = takeStage(true);    // 128->256
  Stage u1 = takeStage(true);    // 256 (+128)->256
  Stage u2 = takeStage(true);    // 256 (+64)->128
  Stage u3 = takeStage(true);    // 128 (+32)->96
  Stage u4 = takeStage(true);    // 96  (+32)->96
  struct Lin { int w, bias, bn; };
  auto takeLin = [&]() { Lin L; L.w = pi++; L.bias = pi++; L.bn = pi; pi += 4; return L; };
  Lin pt0 = takeLin(), pt1 = takeLin(), pt2 = takeLin();

  // ---- workspace bump allocator ----
  size_t off = 0;
  auto alloc  = [&](size_t bytes) {
    size_t a = (off + 255) & ~(size_t)255;
    off = a + bytes;
    return (void*)((char*)d_ws + a);
  };
  auto allocH = [&](long n) { return (h16*)alloc((size_t)n * sizeof(h16)); };
  auto allocF = [&](long n) { return (float*)alloc((size_t)n * sizeof(float)); };

  // ---- buffers ----
  h16* gridIn = allocH(nv0 * 4);
  h16 *m0 = allocH(nv0), *m1 = allocH(nv1), *m2 = allocH(nv2), *m3 = allocH(nv3), *m4 = allocH(nv4);
  float* ACC = allocF(nv0 * 32);    // largest scatter accumulator (stride-1, C=32)
  float* CNT = allocF(nv0);
  h16* A0  = allocH(nv0 * 32);
  h16* X0  = allocH(nv0 * 32);      // skip for u4
  h16* XV  = allocH(nv0 * 32);
  h16* z0  = allocH((long)Np * 32);
  h16* z1  = allocH((long)Np * 256);
  h16* z2  = allocH((long)Np * 128);
  h16* z3  = allocH((long)Np * 96);
  h16* X1  = allocH(nv1 * 32);      // skip for u3
  h16* X2  = allocH(nv2 * 64);      // skip for u2
  h16* X3  = allocH(nv3 * 128);     // skip for u1
  h16* X4  = allocH(nv4 * 256);
  h16* DB  = allocH(nv1 * 32);      // down-conv outputs (largest: 32^3 x 32)
  h16* TMP = allocH(nv0 * 96);      // residual-block conv1 outputs
  h16* SCB = allocH(nv0 * 96);      // shortcut outputs
  h16* RB  = allocH(nv0 * 96);      // residual-block r1 outputs
  h16* CB  = allocH(nv0 * 128);     // concat buffers (largest: 64^3 x 128)
  h16* Y1  = allocH(nv3 * 256);
  h16* Y2  = allocH(nv2 * 128);
  h16* Y3  = allocH(nv1 * 96);
  h16* Y4  = allocH(nv0 * 96);
  h16* YV  = allocH(nv4 * 256);
  h16* YV2 = allocH(nv2 * 128);

  // ---- launch helpers ----
  auto fill0 = [&](float* p, long n) {
    k_fill_f32<<<blks(n), 256, 0, stream>>>(p, n);
  };
  auto prepW = [&](int widx, int Kt, int Cin, int Cout) {
    long n = (long)Kt * Cin * Cout;
    h16* o = allocH(n);
    k_prep_w<<<blks(n), 256, 0, stream>>>(P(widx), o, Kt, Cin, Cout);
    return o;
  };
  auto prepBN = [&](int bnidx, int biasidx, int C, float** sc, float** bi) {
    *sc = allocF(C); *bi = allocF(C);
    k_prep_bn<<<blks(C), 256, 0, stream>>>(P(bnidx), P(bnidx + 1), P(bnidx + 2), P(bnidx + 3),
                                           biasidx >= 0 ? P(biasidx) : nullptr, *sc, *bi, C);
  };
  auto conv = [&](const h16* in, const h16* w, const float* sc, const float* bi,
                  const h16* mask, const h16* resid, h16* out,
                  int g, long M, int Cin, int Cout, int mode, int relu,
                  int istride, int ostride, int ooff, int zdim) {
    dim3 gr((unsigned)((M + 63) / 64), (unsigned)((Cout + 63) / 64), (unsigned)zdim);
    switch (mode) {
      case 0: conv_wmma_t<0><<<gr, 256, 0, stream>>>(in, w, sc, bi, mask, resid, out,
                  g, (int)M, Cin, Cout, relu, istride, ostride, ooff); break;
      case 1: conv_wmma_t<1><<<gr, 256, 0, stream>>>(in, w, sc, bi, mask, resid, out,
                  g, (int)M, Cin, Cout, relu, istride, ostride, ooff); break;
      case 2: conv_wmma_t<2><<<gr, 256, 0, stream>>>(in, w, sc, bi, mask, resid, out,
                  g, (int)M, Cin, Cout, relu, istride, ostride, ooff); break;
      default: conv_wmma_t<3><<<gr, 256, 0, stream>>>(in, w, sc, bi, mask, resid, out,
                  g, (int)M, Cin, Cout, relu, istride, ostride, ooff); break;
    }
  };
  // residual block: relu(bn(c1)*m) -> bn(c2)*m + sc -> relu
  auto resblock = [&](const h16* xin, int istride, int g, long M, int cin, int cout,
                      Layer c1, Layer c2, Layer scl, bool hasSC, const h16* mask, h16* out) {
    float *sA, *bA, *sB, *bB;
    h16* w1 = prepW(c1.w, 27, cin, cout);  prepBN(c1.bn, -1, cout, &sA, &bA);
    h16* w2 = prepW(c2.w, 27, cout, cout); prepBN(c2.bn, -1, cout, &sB, &bB);
    const h16* resid;
    if (hasSC) {
      float *sS, *bS;
      h16* wsc = prepW(scl.w, 1, cin, cout); prepBN(scl.bn, -1, cout, &sS, &bS);
      conv(xin, wsc, sS, bS, mask, nullptr, SCB, g, M, cin, cout, 1, 0, istride, cout, 0, 1);
      resid = SCB;
    } else {
      resid = xin;   // identity shortcut (cin==cout, plain layout)
    }
    conv(xin, w1, sA, bA, mask, nullptr, TMP, g, M, cin, cout, 0, 1, istride, cout, 0, 1);
    conv(TMP, w2, sB, bB, mask, resid, out, g, M, cout, cout, 0, 1, cout, cout, 0, 1);
  };

  // =========================================================================
  // Forward pass
  // =========================================================================
  // initial voxelize: mean point features per voxel, occupancy mask m0
  fill0(ACC, nv0 * 4);
  fill0(CNT, nv0);
  k_vox_init_accum<<<blks(Np), 256, 0, stream>>>(feats, coords, ACC, CNT, Np, G);
  k_vox_init_final<<<blks(nv0), 256, 0, stream>>>(ACC, CNT, gridIn, m0, nv0);
  k_downmask<<<blks(nv1), 256, 0, stream>>>(m0, m1, 32);
  k_downmask<<<blks(nv2), 256, 0, stream>>>(m1, m2, 16);
  k_downmask<<<blks(nv3), 256, 0, stream>>>(m2, m3, 8);
  k_downmask<<<blks(nv4), 256, 0, stream>>>(m3, m4, 4);

  // stem
  { h16* w = prepW(stem0.w, 27, 4, 32); float *s, *b; prepBN(stem0.bn, -1, 32, &s, &b);
    conv(gridIn, w, s, b, m0, nullptr, A0, G, nv0, 4, 32, 0, 1, 4, 32, 0, 1); }
  { h16* w = prepW(stem1.w, 27, 32, 32); float *s, *b; prepBN(stem1.bn, -1, 32, &s, &b);
    conv(A0, w, s, b, m0, nullptr, X0, G, nv0, 32, 32, 0, 1, 32, 32, 0, 1); }

  // z0 = devoxelize(x0);  XV = voxelize(z0)
  k_v2p<<<blks((long)Np * 32), 256, 0, stream>>>(X0, m0, coords, z0, Np, 32, G, 1.f, 0);
  fill0(ACC, nv0 * 32);
  fill0(CNT, nv0);
  k_p2v_accum<<<blks((long)Np * 32), 256, 0, stream>>>(z0, coords, ACC, CNT, Np, 32, G, 1.f);
  k_p2v_final<<<blks(nv0 * 32), 256, 0, stream>>>(ACC, CNT, m0, XV, nv0, 32);

  // encoder stages
  auto encStage = [&](const h16* xin, int go, long Mo, int cin, int cout,
                      Stage S, const h16* mask, h16* Xout) {
    h16* wd = prepW(S.down.w, 8, cin, cin); float *sd, *bd; prepBN(S.down.bn, -1, cin, &sd, &bd);
    conv(xin, wd, sd, bd, mask, nullptr, DB, go, Mo, cin, cin, 2, 1, cin, cin, 0, 1);
    resblock(DB, cin, go, Mo, cin, cout, S.r1c1, S.r1c2, S.r1sc, S.hasSC, mask, RB);
    resblock(RB, cout, go, Mo, cout, cout, S.r2c1, S.r2c2, S.r1sc, false, mask, Xout);
  };
  encStage(XV, 32, nv1, 32, 32, s1, m1, X1);
  encStage(X1, 16, nv2, 32, 64, s2, m2, X2);
  encStage(X2, 8,  nv3, 64, 128, s3, m3, X3);
  encStage(X3, 4,  nv4, 128, 256, s4, m4, X4);

  // z1 = pt0(z0) + devoxelize(x4);  YV = voxelize(z1, stride 16)
  { h16* w = prepW(pt0.w, 1, 32, 256); float *s, *b; prepBN(pt0.bn, pt0.bias, 256, &s, &b);
    conv(z0, w, s, b, nullptr, nullptr, z1, 1, Np, 32, 256, 1, 1, 32, 256, 0, 1); }
  k_v2p<<<blks((long)Np * 256), 256, 0, stream>>>(X4, m4, coords, z1, Np, 256, 4, 1.f / 16.f, 1);
  fill0(ACC, nv4 * 256);
  fill0(CNT, nv4);
  k_p2v_accum<<<blks((long)Np * 256), 256, 0, stream>>>(z1, coords, ACC, CNT, Np, 256, 4, 1.f / 16.f);
  k_p2v_final<<<blks(nv4 * 256), 256, 0, stream>>>(ACC, CNT, m4, YV, nv4, 256);

  // decoder stages (2x deconv via 8 parity GEMMs -> concat -> 2 residual blocks)
  auto decStage = [&](const h16* yin, int gc, const h16* skip, int cskip,
                      int cin, int cout, Stage U, const h16* maskF, long Mf, h16* Yout) {
    int gf = 2 * gc;
    int ctot = cout + cskip;
    long Mc = (long)gc * gc * gc;
    h16* wd = prepW(U.down.w, 8, cin, cout); float *sd, *bd; prepBN(U.down.bn, -1, cout, &sd, &bd);
    conv(yin, wd, sd, bd, maskF, nullptr, CB, gc, Mc, cin, cout, 3, 1, cin, ctot, 0, 8);
    k_copy_ch<<<blks(Mf * cskip), 256, 0, stream>>>(skip, CB, Mf, cskip, cskip, 0, ctot, cout);
    resblock(CB, ctot, gf, Mf, ctot, cout, U.r1c1, U.r1c2, U.r1sc, true, maskF, RB);
    resblock(RB, cout, gf, Mf, cout, cout, U.r2c1, U.r2c2, U.r1sc, false, maskF, Yout);
  };
  decStage(YV, 4,  X3, 128, 256, 256, u1, m3, nv3, Y1);
  decStage(Y1, 8,  X2, 64,  256, 128, u2, m2, nv2, Y2);

  // z2 = pt1(z1) + devoxelize(y2);  YV2 = voxelize(z2, stride 4)
  { h16* w = prepW(pt1.w, 1, 256, 128); float *s, *b; prepBN(pt1.bn, pt1.bias, 128, &s, &b);
    conv(z1, w, s, b, nullptr, nullptr, z2, 1, Np, 256, 128, 1, 1, 256, 128, 0, 1); }
  k_v2p<<<blks((long)Np * 128), 256, 0, stream>>>(Y2, m2, coords, z2, Np, 128, 16, 0.25f, 1);
  fill0(ACC, nv2 * 128);
  fill0(CNT, nv2);
  k_p2v_accum<<<blks((long)Np * 128), 256, 0, stream>>>(z2, coords, ACC, CNT, Np, 128, 16, 0.25f);
  k_p2v_final<<<blks(nv2 * 128), 256, 0, stream>>>(ACC, CNT, m2, YV2, nv2, 128);

  decStage(YV2, 16, X1, 32, 128, 96, u3, m1, nv1, Y3);
  decStage(Y3,  32, X0, 32, 96,  96, u4, m0, nv0, Y4);

  // z3 = pt2(z2) + devoxelize(y4); output f32
  { h16* w = prepW(pt2.w, 1, 128, 96); float *s, *b; prepBN(pt2.bn, pt2.bias, 96, &s, &b);
    conv(z2, w, s, b, nullptr, nullptr, z3, 1, Np, 128, 96, 1, 1, 128, 96, 0, 1); }
  k_v2p<<<blks((long)Np * 96), 256, 0, stream>>>(Y4, m0, coords, z3, Np, 96, G, 1.f, 1);
  k_h2f<<<blks((long)Np * 96), 256, 0, stream>>>(z3, (float*)Np == nullptr ? (float*)d_out : (float*)d_out, (long)Np * 96);
}
